// TaskConditionedEGNNLayer_3152505995605
// MI455X (gfx1250) — matrix operand order repaired and checked by reading, not compiled, not run on hardware
//
#include <hip/hip_runtime.h>

typedef _Float16 v16h __attribute__((ext_vector_type(16)));
typedef _Float16 v8h  __attribute__((ext_vector_type(8)));
typedef _Float16 v4h  __attribute__((ext_vector_type(4)));
typedef float    v8f  __attribute__((ext_vector_type(8)));

#define HID 64
#define KE  192   // edge-input K, 177 padded to 6*32 (col 177 = 1.0 bias row)
#define KN  160   // node-input K = 5*32

// Permuted edge-input column map (weights permuted identically in prep_ew1):
//   [0,64)   h[row]      [64,128) h[col]     [128,144) edge_attr
//   [144,176) task_emb[row]   [176] dist   [177] 1.0 (bias)   [178,192) 0

__device__ __forceinline__ float silu_f(float x) { return x / (1.0f + __expf(-x)); }

__device__ __forceinline__ v8f splat8(float x) {
    v8f v;
#pragma unroll
    for (int i = 0; i < 8; ++i) v[i] = x;
    return v;
}

// Convert a float4 to 4 packed halfs and store as one 8-byte LDS store.
__device__ __forceinline__ void st4h(_Float16* p, float4 v) {
    v4h q;
    q[0] = (_Float16)v.x; q[1] = (_Float16)v.y;
    q[2] = (_Float16)v.z; q[3] = (_Float16)v.w;
    *(v4h*)p = q;
}

// A-fragment in this file's contiguous k order (B prep matches; not AMD's table order): lanes 0-15 hold M=lane, K = kt*32 + {0..15};
// lanes 16-31 hold M=lane-16, K = kt*32 + {16..31}.
__device__ __forceinline__ v16h load_a_frag(const _Float16* base, int stride, int kt, int lane) {
    const int m  = lane & 15;
    const int hi = lane >> 4;
    const _Float16* p = base + m * stride + kt * 32 + hi * 16;
    v8h lo = *(const v8h*)(p);
    v8h hh = *(const v8h*)(p + 8);
    return __builtin_shufflevector(lo, hh, 0,1,2,3,4,5,6,7,8,9,10,11,12,13,14,15);
}

// B-fragments pre-tiled by prep kernels: 16 contiguous halfs per lane per fragment.
__device__ __forceinline__ v16h load_b_frag(const _Float16* frags, int frag, int lane) {
    return *(const v16h*)(frags + (size_t)(frag * 32 + lane) * 16);
}

#define WMMA_F16(a, b, c) \
    __builtin_amdgcn_wmma_f32_16x16x32_f16(false, (a), false, (b), (short)0, (c), false, false)

// Same-wave LDS phase fence: HW keeps one wave's DS ops in order; this stops the
// compiler reordering across phases and drains DScnt.
#define LDS_FENCE() asm volatile("s_wait_dscnt 0" ::: "memory")

// ---------------------------------------------------------------------------
// Generic [Kreal x 64] f32 -> fragment-major f16. grid = ktiles*4, 32 threads.
// ---------------------------------------------------------------------------
__global__ void prep_weights(const float* __restrict__ W, int Kreal, _Float16* __restrict__ dst) {
    const int frag = blockIdx.x;
    const int kt = frag >> 2, nt = frag & 3;
    const int lane = threadIdx.x;
    const int n = nt * 16 + (lane & 15);
    const int kbase = kt * 32 + (lane >> 4) * 16;
#pragma unroll
    for (int j = 0; j < 16; ++j) {
        const int k = kbase + j;
        const float v = (k < Kreal) ? W[(size_t)k * 64 + n] : 0.0f;
        dst[((size_t)frag * 32 + lane) * 16 + j] = (_Float16)v;
    }
}

// ---------------------------------------------------------------------------
// ew1 with K-permutation (task_emb<->dist swap) and bias folded at row 177.
// ---------------------------------------------------------------------------
__global__ void prep_ew1(const float* __restrict__ W, const float* __restrict__ bias,
                         _Float16* __restrict__ dst) {
    const int frag = blockIdx.x;
    const int kt = frag >> 2, nt = frag & 3;
    const int lane = threadIdx.x;
    const int n = nt * 16 + (lane & 15);
    const int kbase = kt * 32 + (lane >> 4) * 16;
#pragma unroll
    for (int j = 0; j < 16; ++j) {
        const int k = kbase + j;
        float v;
        if (k < 144)       v = W[(size_t)k * 64 + n];            // h_row | h_col | edge_attr
        else if (k < 176)  v = W[(size_t)(k + 1) * 64 + n];      // task_emb (orig rows 145..176)
        else if (k == 176) v = W[(size_t)144 * 64 + n];          // dist (orig row 144)
        else if (k == 177) v = bias[n];                          // folded eb1
        else               v = 0.0f;
        dst[((size_t)frag * 32 + lane) * 16 + j] = (_Float16)v;
    }
}

// ---------------------------------------------------------------------------
// Zero the m_i accumulator; seed pos_out with pos (edge kernel accumulates).
// ---------------------------------------------------------------------------
__global__ void init_accum(float* __restrict__ m_i, int n_mi,
                           const float* __restrict__ pos, float* __restrict__ pos_out, int n_pos) {
    const int stride = gridDim.x * blockDim.x;
    for (int i = blockIdx.x * blockDim.x + threadIdx.x; i < n_mi; i += stride) m_i[i] = 0.0f;
    for (int i = blockIdx.x * blockDim.x + threadIdx.x; i < n_pos; i += stride) pos_out[i] = pos[i];
}

// ---------------------------------------------------------------------------
// Edge kernel: one wave per 16-edge tile.
// ---------------------------------------------------------------------------
__global__ __launch_bounds__(256) void edge_kernel(
    const float* __restrict__ h, const float* __restrict__ pos,
    const float* __restrict__ edge_attr, const float* __restrict__ task_emb,
    const int* __restrict__ edge_index,
    const _Float16* __restrict__ wE1,
    const _Float16* __restrict__ wE2, const float* __restrict__ eb2,
    const _Float16* __restrict__ wC1, const float* __restrict__ cb1,
    const float* __restrict__ cw2,
    float* __restrict__ m_i, float* __restrict__ pos_out, int Ee) {
    __shared__ __align__(64) _Float16 sBuf[8][16 * KE];   // per-wave staging / y1 / m_ij / z
    __shared__ int sCol[8][16];

    const int lane = threadIdx.x & 31;
    const int wid  = threadIdx.x >> 5;
    const int m    = lane & 15;
    const int hi   = lane >> 4;
    const int T    = (Ee + 15) >> 4;
    _Float16* A = sBuf[wid];

    for (int t = blockIdx.x * 8 + wid; t < T; t += gridDim.x * 8) {
        const int e     = t * 16 + m;
        const bool valid = (e < Ee);
        const int ec = valid ? e : 0;
        const int r  = edge_index[ec];        // row
        const int c  = edge_index[Ee + ec];   // col
        if (hi == 0) sCol[wid][m] = c;

        const float px = pos[r * 3 + 0] - pos[c * 3 + 0];
        const float py = pos[r * 3 + 1] - pos[c * 3 + 1];
        const float pz = pos[r * 3 + 2] - pos[c * 3 + 2];
        const float dist = sqrtf(px * px + py * py + pz * pz);

        // Vectorized staging: b128 gathers -> packed f16 -> b64 LDS stores.
        {
            _Float16* row = A + m * KE;
            const float4* hr4 = (const float4*)(h + (size_t)r * 64);
            const float4* hc4 = (const float4*)(h + (size_t)c * 64);
            if (hi == 0) {
#pragma unroll
                for (int q = 0; q < 16; ++q) st4h(row + q * 4, hr4[q]);
#pragma unroll
                for (int q = 0; q < 8; ++q)  st4h(row + 64 + q * 4, hc4[q]);
            } else {
#pragma unroll
                for (int q = 0; q < 8; ++q)  st4h(row + 96 + q * 4, hc4[8 + q]);
                const float4* ea4 = (const float4*)(edge_attr + (size_t)ec * 16);
#pragma unroll
                for (int q = 0; q < 4; ++q)  st4h(row + 128 + q * 4, ea4[q]);
                const float4* te4 = (const float4*)(task_emb + (size_t)r * 32);
#pragma unroll
                for (int q = 0; q < 8; ++q)  st4h(row + 144 + q * 4, te4[q]);
                st4h(row + 176, make_float4(dist, 1.0f, 0.0f, 0.0f));
                st4h(row + 180, make_float4(0.0f, 0.0f, 0.0f, 0.0f));
                st4h(row + 184, make_float4(0.0f, 0.0f, 0.0f, 0.0f));
                st4h(row + 188, make_float4(0.0f, 0.0f, 0.0f, 0.0f));
            }
        }
        LDS_FENCE();

        // Layer 1: [16x192] @ [192x64]; bias folded into weight row 177.
        v8f acc[4];
#pragma unroll
        for (int n0 = 0; n0 < 4; ++n0) acc[n0] = splat8(0.0f);
#pragma unroll
        for (int kt = 0; kt < 6; ++kt) {
            const v16h a = load_a_frag(A, KE, kt, lane);
#pragma unroll
            for (int n0 = 0; n0 < 4; ++n0)
                acc[n0] = WMMA_F16(a, load_b_frag(wE1, kt * 4 + n0, lane), acc[n0]);
        }
        LDS_FENCE();
        _Float16* Y = A;  // reuse staging as [16][64] f16
#pragma unroll
        for (int n0 = 0; n0 < 4; ++n0)
#pragma unroll
            for (int rr = 0; rr < 8; ++rr) {
                const int M = rr + 8 * hi;
                Y[M * 64 + n0 * 16 + m] = (_Float16)silu_f(acc[n0][rr]);
            }
        LDS_FENCE();

        // Layer 2: m_ij = silu([16x64] @ [64x64] + eb2).
        v8f mc[4];
#pragma unroll
        for (int n0 = 0; n0 < 4; ++n0) mc[n0] = splat8(eb2[n0 * 16 + m]);
#pragma unroll
        for (int kt = 0; kt < 2; ++kt) {
            const v16h a = load_a_frag(Y, 64, kt, lane);
#pragma unroll
            for (int n0 = 0; n0 < 4; ++n0)
                mc[n0] = WMMA_F16(a, load_b_frag(wE2, kt * 4 + n0, lane), mc[n0]);
        }
        LDS_FENCE();
        // Scatter m_ij into m_i[col] and restage m_ij for the coord GEMM.
#pragma unroll
        for (int n0 = 0; n0 < 4; ++n0)
#pragma unroll
            for (int rr = 0; rr < 8; ++rr) {
                const int M = rr + 8 * hi;
                const float v = silu_f(mc[n0][rr]);
                Y[M * 64 + n0 * 16 + m] = (_Float16)v;
                const int ee = t * 16 + M;
                if (ee < Ee)
                    atomicAdd(&m_i[(size_t)sCol[wid][M] * 64 + n0 * 16 + m], v);
            }
        LDS_FENCE();

        // Coord path: z = silu(m_ij @ cw1 + cb1).
        v8f zc[4];
#pragma unroll
        for (int n0 = 0; n0 < 4; ++n0) zc[n0] = splat8(cb1[n0 * 16 + m]);
#pragma unroll
        for (int kt = 0; kt < 2; ++kt) {
            const v16h a = load_a_frag(Y, 64, kt, lane);
#pragma unroll
            for (int n0 = 0; n0 < 4; ++n0)
                zc[n0] = WMMA_F16(a, load_b_frag(wC1, kt * 4 + n0, lane), zc[n0]);
        }
        LDS_FENCE();
        _Float16* Z = A + 1024;  // [16][64] f16, disjoint from Y
#pragma unroll
        for (int n0 = 0; n0 < 4; ++n0)
#pragma unroll
            for (int rr = 0; rr < 8; ++rr) {
                const int M = rr + 8 * hi;
                Z[M * 64 + n0 * 16 + m] = (_Float16)silu_f(zc[n0][rr]);
            }
        LDS_FENCE();

        // coord_weights = z @ cw2 ; pos_out[col] += w * rel / (dist + 1e-8).
        if (hi == 0 && valid) {
            float w = 0.0f;
#pragma unroll
            for (int k = 0; k < 64; k += 4) {
                const v4h z4 = *(const v4h*)(Z + m * 64 + k);
                const float4 c4 = *(const float4*)(cw2 + k);
                w += (float)z4[0] * c4.x + (float)z4[1] * c4.y +
                     (float)z4[2] * c4.z + (float)z4[3] * c4.w;
            }
            const float s = w / (dist + 1e-8f);
            atomicAdd(&pos_out[(size_t)c * 3 + 0], px * s);
            atomicAdd(&pos_out[(size_t)c * 3 + 1], py * s);
            atomicAdd(&pos_out[(size_t)c * 3 + 2], pz * s);
        }
        LDS_FENCE();
    }
}

// ---------------------------------------------------------------------------
// Node kernel: node MLP + FiLM + residual, one wave per 16-node tile.
// ---------------------------------------------------------------------------
__global__ __launch_bounds__(256) void node_kernel(
    const float* __restrict__ h, const float* __restrict__ task_emb,
    const float* __restrict__ m_i,
    const _Float16* __restrict__ wN1, const float* __restrict__ nb1,
    const _Float16* __restrict__ wN2, const float* __restrict__ nb2,
    const _Float16* __restrict__ wS, const float* __restrict__ sb,
    const _Float16* __restrict__ wT, const float* __restrict__ tb,
    float* __restrict__ h_out, int Nn) {
    __shared__ __align__(64) _Float16 sBuf[8][16 * KN + 16 * 64];  // staging + u

    const int lane = threadIdx.x & 31;
    const int wid  = threadIdx.x >> 5;
    const int m    = lane & 15;
    const int hi   = lane >> 4;
    const int T    = (Nn + 15) >> 4;
    _Float16* A = sBuf[wid];
    _Float16* U = A + 16 * KN;

    for (int t = blockIdx.x * 8 + wid; t < T; t += gridDim.x * 8) {
        const int node = t * 16 + m;
        const int nc   = (node < Nn) ? node : 0;

        // Vectorized staging: cols [0,64)=h, [64,128)=m_i, [128,160)=task_emb.
        {
            _Float16* row = A + m * KN;
            const float4* h4  = (const float4*)(h + (size_t)nc * 64);
            const float4* mi4 = (const float4*)(m_i + (size_t)nc * 64);
            if (hi == 0) {
#pragma unroll
                for (int q = 0; q < 16; ++q) st4h(row + q * 4, h4[q]);
#pragma unroll
                for (int q = 0; q < 4; ++q)  st4h(row + 64 + q * 4, mi4[q]);
            } else {
#pragma unroll
                for (int q = 0; q < 12; ++q) st4h(row + 80 + q * 4, mi4[4 + q]);
                const float4* te4 = (const float4*)(task_emb + (size_t)nc * 32);
#pragma unroll
                for (int q = 0; q < 8; ++q)  st4h(row + 128 + q * 4, te4[q]);
            }
        }
        LDS_FENCE();

        // Layer 1: u = silu([16x160] @ nw1 + nb1).
        v8f acc[4];
#pragma unroll
        for (int n0 = 0; n0 < 4; ++n0) acc[n0] = splat8(nb1[n0 * 16 + m]);
#pragma unroll
        for (int kt = 0; kt < 5; ++kt) {
            const v16h a = load_a_frag(A, KN, kt, lane);
#pragma unroll
            for (int n0 = 0; n0 < 4; ++n0)
                acc[n0] = WMMA_F16(a, load_b_frag(wN1, kt * 4 + n0, lane), acc[n0]);
        }
        LDS_FENCE();
#pragma unroll
        for (int n0 = 0; n0 < 4; ++n0)
#pragma unroll
            for (int rr = 0; rr < 8; ++rr) {
                const int M = rr + 8 * hi;
                U[M * 64 + n0 * 16 + m] = (_Float16)silu_f(acc[n0][rr]);
            }
        LDS_FENCE();

        // FiLM A-fragment is node_input columns 128..159 (kt = 4 over stride KN).
        const v16h aT = load_a_frag(A, KN, 4, lane);

#pragma unroll
        for (int n0 = 0; n0 < 4; ++n0) {
            v8f hc = splat8(nb2[n0 * 16 + m]);   // u @ nw2 + nb2
#pragma unroll
            for (int kt = 0; kt < 2; ++kt)
                hc = WMMA_F16(load_a_frag(U, 64, kt, lane),
                              load_b_frag(wN2, kt * 4 + n0, lane), hc);
            v8f sc = splat8(sb[n0 * 16 + m]);    // task_emb @ sw + sb
            sc = WMMA_F16(aT, load_b_frag(wS, n0, lane), sc);
            v8f tc = splat8(tb[n0 * 16 + m]);    // task_emb @ tw + tb
            tc = WMMA_F16(aT, load_b_frag(wT, n0, lane), tc);

#pragma unroll
            for (int rr = 0; rr < 8; ++rr) {
                const int M = rr + 8 * hi;
                const int nn = t * 16 + M;
                if (nn < Nn) {
                    const size_t idx = (size_t)nn * 64 + n0 * 16 + m;
                    h_out[idx] = h[idx] + sc[rr] * hc[rr] + tc[rr];
                }
            }
        }
        LDS_FENCE();
    }
}

// ---------------------------------------------------------------------------
extern "C" void kernel_launch(void* const* d_in, const int* in_sizes, int n_in,
                              void* d_out, int out_size, void* d_ws, size_t ws_size,
                              hipStream_t stream) {
    const float* h         = (const float*)d_in[0];
    const float* pos       = (const float*)d_in[1];
    const float* edge_attr = (const float*)d_in[2];
    const float* task_emb  = (const float*)d_in[3];
    const int*   edge_index= (const int*)d_in[4];
    const float* ew1 = (const float*)d_in[5];
    const float* eb1 = (const float*)d_in[6];
    const float* ew2 = (const float*)d_in[7];
    const float* eb2 = (const float*)d_in[8];
    const float* nw1 = (const float*)d_in[9];
    const float* nb1 = (const float*)d_in[10];
    const float* nw2 = (const float*)d_in[11];
    const float* nb2 = (const float*)d_in[12];
    const float* cw1 = (const float*)d_in[13];
    const float* cb1 = (const float*)d_in[14];
    const float* cw2 = (const float*)d_in[15];
    const float* sw  = (const float*)d_in[16];
    const float* sb  = (const float*)d_in[17];
    const float* tw  = (const float*)d_in[18];
    const float* tb  = (const float*)d_in[19];
    (void)n_in; (void)out_size; (void)ws_size;

    const int Nn = in_sizes[0] / 64;   // 100000
    const int Ee = in_sizes[2] / 16;   // 1600000

    // Workspace layout: [m_i : Nn*64 f32][f16 weight fragments].
    float* m_i = (float*)d_ws;
    const size_t miBytes = (((size_t)Nn * 64 * sizeof(float)) + 1023) & ~(size_t)1023;
    _Float16* wE1 = (_Float16*)((char*)d_ws + miBytes);       // 24 frags
    _Float16* wE2 = wE1 + 24 * 512;                           // 8
    _Float16* wN1 = wE2 + 8 * 512;                            // 20
    _Float16* wN2 = wN1 + 20 * 512;                           // 8
    _Float16* wC1 = wN2 + 8 * 512;                            // 8
    _Float16* wS  = wC1 + 8 * 512;                            // 4
    _Float16* wT  = wS  + 4 * 512;                            // 4

    float* h_out   = (float*)d_out;
    float* pos_out = h_out + (size_t)Nn * 64;

    init_accum<<<1024, 256, 0, stream>>>(m_i, Nn * 64, pos, pos_out, Nn * 3);
    prep_ew1<<<24, 32, 0, stream>>>(ew1, eb1, wE1);
    prep_weights<<<8,  32, 0, stream>>>(ew2, 64,  wE2);
    prep_weights<<<20, 32, 0, stream>>>(nw1, 160, wN1);
    prep_weights<<<8,  32, 0, stream>>>(nw2, 64,  wN2);
    prep_weights<<<8,  32, 0, stream>>>(cw1, 64,  wC1);
    prep_weights<<<4,  32, 0, stream>>>(sw,  32,  wS);
    prep_weights<<<4,  32, 0, stream>>>(tw,  32,  wT);

    const int Tt = (Ee + 15) / 16;
    int eBlocks = (Tt + 7) / 8; if (eBlocks > 3072) eBlocks = 3072;
    edge_kernel<<<eBlocks, 256, 0, stream>>>(h, pos, edge_attr, task_emb, edge_index,
                                             wE1, wE2, eb2, wC1, cb1, cw2,
                                             m_i, pos_out, Ee);

    const int Tn = (Nn + 15) / 16;
    int nBlocks = (Tn + 7) / 8; if (nBlocks > 1024) nBlocks = 1024;
    node_kernel<<<nBlocks, 256, 0, stream>>>(h, task_emb, m_i,
                                             wN1, nb1, wN2, nb2, wS, sb, wT, tb,
                                             h_out, Nn);
}
